// GATPeptideStruct_20461224198770
// MI455X (gfx1250) — matrix operand order repaired and checked by reading, not compiled, not run on hardware
//
#include <hip/hip_runtime.h>

// ---------------------------------------------------------------------------
// GAT (3 layers, heads={4,4,1}) + mean pool + FC for MI455X (gfx1250).
// Dense GEMMs use v_wmma_f32_16x16x32_bf16 (wave32 WMMA). Edge softmax +
// aggregation use fp32 global atomics. Transformed features kept in bf16 to
// halve the gather bandwidth of the edge phase (the roofline bottleneck).
// ---------------------------------------------------------------------------

typedef __attribute__((ext_vector_type(16))) __bf16 v16bf;
typedef __attribute__((ext_vector_type(8)))  float  v8f;

#define N_NODES  50000
#define N_EDGES  800000
#define N_GRAPHS 128
#define NEG_SLOPE 0.2f
#define ET (N_EDGES + N_NODES)   // edges incl. self loops

static inline int cdiv(int a, int b) { return (a + b - 1) / b; }

// ---------------------------------------------------------------------------
// utility kernels
// ---------------------------------------------------------------------------
__global__ void fill_f32_kernel(float* p, float v, size_t n) {
    size_t i = (size_t)blockIdx.x * blockDim.x + threadIdx.x;
    if (i < n) p[i] = v;
}

// Wt[n*K + k] = (bf16) W[k*Nout + n]   (transpose + convert, W stays L2 resident)
__global__ void convert_w_kernel(const float* __restrict__ W, __bf16* __restrict__ Wt,
                                 int K, int Nout) {
    int i = blockIdx.x * blockDim.x + threadIdx.x;
    if (i >= K * Nout) return;
    int k = i % K, n = i / K;
    Wt[(size_t)n * K + k] = (__bf16)W[(size_t)k * Nout + n];
}

// ---------------------------------------------------------------------------
// WMMA GEMM:  G[N, Nout] (bf16) = A[N, K] (f32) x Wt[Nout, K] (bf16, K-major)
// block = 256 threads = 8 waves; block tile 128(M) x 64(N); wave tile 16 x 64.
// K in {128,256,512} (multiple of 32). Rows clamped on load, masked on store.
// ---------------------------------------------------------------------------
__global__ void __launch_bounds__(256)
gemm_wmma_bf16_kernel(const float* __restrict__ A, const __bf16* __restrict__ Wt,
                      __bf16* __restrict__ G, int Nrows, int K, int Nout) {
    const int wave  = threadIdx.x >> 5;
    const int lane  = threadIdx.x & 31;
    const int l16   = lane & 15;
    const int khalf = lane >> 4;            // 0 or 1

    const int mBase = blockIdx.x * 128 + wave * 16;   // wave's 16-row strip
    const int nBase = blockIdx.y * 64;

    // A row for this lane's fragment (clamped; out-of-range rows are masked at store)
    int rowA = mBase + l16;
    if (rowA >= Nrows) rowA = Nrows - 1;
    const float* __restrict__ Arow = A + (size_t)rowA * K;

    v8f acc[4] = {};

    for (int ko = 0; ko < K; ko += 32) {
        // ---- A fragment (16x32 bf16), contiguous order matching B:
        // lane<16 : elems 0..15 -> K=ko+0..15
        // lane>=16: elems 0..15 -> K=ko+16..31
        const float4* pa0 = (const float4*)(Arow + ko + khalf * 16);
        const float4* pa1 = (const float4*)(Arow + ko + khalf * 16 + 8);
        float4 a0 = pa0[0], a1 = pa0[1];
        float4 a2 = pa1[0], a3 = pa1[1];
        if (ko + 32 < K) __builtin_prefetch((const void*)(Arow + ko + 32), 0, 1);

        v16bf av;
        av[0]  = (__bf16)a0.x; av[1]  = (__bf16)a0.y; av[2]  = (__bf16)a0.z; av[3]  = (__bf16)a0.w;
        av[4]  = (__bf16)a1.x; av[5]  = (__bf16)a1.y; av[6]  = (__bf16)a1.z; av[7]  = (__bf16)a1.w;
        av[8]  = (__bf16)a2.x; av[9]  = (__bf16)a2.y; av[10] = (__bf16)a2.z; av[11] = (__bf16)a2.w;
        av[12] = (__bf16)a3.x; av[13] = (__bf16)a3.y; av[14] = (__bf16)a3.z; av[15] = (__bf16)a3.w;

        // ---- B fragments: lane holds column N=l16, K = ko + khalf*16 + 0..15
        // Wt is K-major per column -> one contiguous 32B load per fragment.
#pragma unroll
        for (int n = 0; n < 4; ++n) {
            const int col = nBase + n * 16 + l16;
            const v16bf bv = *(const v16bf*)(Wt + (size_t)col * K + ko + khalf * 16);
            acc[n] = __builtin_amdgcn_wmma_f32_16x16x32_bf16(
                false, av, false, bv, (short)0, acc[n], false, false);
        }
    }

    // ---- store D (lane l, vgpr r): row = mBase + (lane>>4)*8 + r, col = nBase+n*16+l16
#pragma unroll
    for (int n = 0; n < 4; ++n) {
        const int col = nBase + n * 16 + l16;
#pragma unroll
        for (int r = 0; r < 8; ++r) {
            const int row = mBase + khalf * 8 + r;
            if (row < Nrows) G[(size_t)row * Nout + col] = (__bf16)acc[n][r];
        }
    }
}

// ---------------------------------------------------------------------------
// alpha_s / alpha_d per (node, head): wave32 reduction over channels
// ---------------------------------------------------------------------------
__global__ void alpha_kernel(const __bf16* __restrict__ g,
                             const float* __restrict__ a_src, const float* __restrict__ a_dst,
                             float* __restrict__ as, float* __restrict__ ad,
                             int N, int H, int C) {
    const int wid  = (int)((blockIdx.x * (size_t)blockDim.x + threadIdx.x) >> 5);
    const int lane = threadIdx.x & 31;
    if (wid >= N * H) return;
    const int node = wid / H, h = wid % H;
    const __bf16* row = g + (size_t)node * H * C + (size_t)h * C;
    const float* vs = a_src + h * C;
    const float* vd = a_dst + h * C;
    float ss = 0.f, sd = 0.f;
    for (int c = lane; c < C; c += 32) {
        float x = (float)row[c];
        ss += x * vs[c];
        sd += x * vd[c];
    }
#pragma unroll
    for (int off = 16; off; off >>= 1) {
        ss += __shfl_xor(ss, off, 32);
        sd += __shfl_xor(sd, off, 32);
    }
    if (lane == 0) { as[wid] = ss; ad[wid] = sd; }
}

// ---------------------------------------------------------------------------
// edge softmax passes
// ---------------------------------------------------------------------------
__device__ inline void atomicMaxF(float* addr, float val) {
    // ordered-int encoding: correct for mixed-sign floats
    if (val >= 0.f) atomicMax((int*)addr, __float_as_int(val));
    else            atomicMin((unsigned int*)addr, __float_as_uint(val));
}

__global__ void edge_logits_kernel(const float* __restrict__ as, const float* __restrict__ ad,
                                   const long long* __restrict__ ei,
                                   float* __restrict__ ebuf, float* __restrict__ m, int H) {
    int e = blockIdx.x * blockDim.x + threadIdx.x;
    if (e >= ET) return;
    int src, dst;
    if (e < N_EDGES) { src = (int)ei[e]; dst = (int)ei[N_EDGES + e]; }
    else             { src = dst = e - N_EDGES; }
    for (int h = 0; h < H; ++h) {
        float v = as[(size_t)src * H + h] + ad[(size_t)dst * H + h];
        v = (v > 0.f) ? v : NEG_SLOPE * v;                 // leaky relu
        ebuf[(size_t)e * H + h] = v;
        atomicMaxF(m + (size_t)dst * H + h, v);
    }
}

__global__ void edge_exp_kernel(const long long* __restrict__ ei,
                                const float* __restrict__ m,
                                float* __restrict__ ebuf, float* __restrict__ denom, int H) {
    int e = blockIdx.x * blockDim.x + threadIdx.x;
    if (e >= ET) return;
    int dst = (e < N_EDGES) ? (int)ei[N_EDGES + e] : (e - N_EDGES);
    for (int h = 0; h < H; ++h) {
        float ex = __expf(ebuf[(size_t)e * H + h] - m[(size_t)dst * H + h]);
        ebuf[(size_t)e * H + h] = ex;
        atomicAdd(denom + (size_t)dst * H + h, ex);
    }
}

// one block per edge: out[dst] += (e'/denom[dst]) * h[src]   (fp32 atomic scatter)
__global__ void edge_aggr_kernel(const __bf16* __restrict__ g,
                                 const float* __restrict__ ebuf, const float* __restrict__ denom,
                                 const long long* __restrict__ ei,
                                 float* __restrict__ out, int H, int C) {
    int e = blockIdx.x;
    int src, dst;
    if (e < N_EDGES) { src = (int)ei[e]; dst = (int)ei[N_EDGES + e]; }
    else             { src = dst = e - N_EDGES; }
    const int HC = H * C;
    for (int idx = threadIdx.x; idx < HC; idx += blockDim.x) {
        int h = idx / C;
        float w = ebuf[(size_t)e * H + h] / denom[(size_t)dst * H + h];
        float v = w * (float)g[(size_t)src * HC + idx];
        atomicAdd(out + (size_t)dst * HC + idx, v);
    }
}

__global__ void bias_relu_kernel(float* __restrict__ p, const float* __restrict__ bias,
                                 size_t n, int HC) {
    size_t i = (size_t)blockIdx.x * blockDim.x + threadIdx.x;
    if (i >= n) return;
    float v = p[i] + bias[i % HC];
    p[i] = v > 0.f ? v : 0.f;
}

// ---------------------------------------------------------------------------
// pooling + FC
// ---------------------------------------------------------------------------
__global__ void pool_kernel(const float* __restrict__ h3, const long long* __restrict__ batch,
                            float* __restrict__ sums, float* __restrict__ counts) {
    int node = blockIdx.x;
    int c = threadIdx.x;                      // block = 64 threads
    int b = (int)batch[node];
    atomicAdd(&sums[(size_t)b * 64 + c], h3[(size_t)node * 64 + c]);
    if (c == 0) atomicAdd(&counts[b], 1.0f);
}

__global__ void fc_kernel(const float* __restrict__ sums, const float* __restrict__ counts,
                          const float* __restrict__ Wfc, const float* __restrict__ bfc,
                          float* __restrict__ out) {
    int i = blockIdx.x * blockDim.x + threadIdx.x;
    if (i >= N_GRAPHS * 10) return;
    int gidx = i / 10, j = i % 10;
    float cnt = counts[gidx];
    cnt = cnt > 1.f ? cnt : 1.f;
    float acc = 0.f;
    for (int c = 0; c < 64; ++c)
        acc += (sums[(size_t)gidx * 64 + c] / cnt) * Wfc[(size_t)c * 10 + j];
    out[i] = acc + bfc[j];
}

// ---------------------------------------------------------------------------
// host side
// ---------------------------------------------------------------------------
extern "C" void kernel_launch(void* const* d_in, const int* in_sizes, int n_in,
                              void* d_out, int out_size, void* d_ws, size_t ws_size,
                              hipStream_t stream) {
    (void)in_sizes; (void)n_in; (void)out_size; (void)ws_size;

    const float*      x      = (const float*)d_in[0];
    const long long*  ei     = (const long long*)d_in[1];
    const long long*  batch  = (const long long*)d_in[2];
    const float*      W1     = (const float*)d_in[3];
    const float*      as1    = (const float*)d_in[4];
    const float*      ad1    = (const float*)d_in[5];
    const float*      b1     = (const float*)d_in[6];
    const float*      W2     = (const float*)d_in[7];
    const float*      as2    = (const float*)d_in[8];
    const float*      ad2    = (const float*)d_in[9];
    const float*      b2     = (const float*)d_in[10];
    const float*      W3     = (const float*)d_in[11];
    const float*      as3    = (const float*)d_in[12];
    const float*      ad3    = (const float*)d_in[13];
    const float*      b3     = (const float*)d_in[14];
    const float*      Wfc    = (const float*)d_in[15];
    const float*      bfc    = (const float*)d_in[16];
    float*            out    = (float*)d_out;

    // workspace carve-up (256B aligned)
    char* ws = (char*)d_ws;
    size_t off = 0;
    auto alloc = [&](size_t bytes) -> void* {
        void* p = ws + off;
        off += (bytes + 255) & ~(size_t)255;
        return p;
    };
    float*   f_buf  = (float*)  alloc((size_t)N_NODES * 512 * 4); // h2
    float*   o_buf  = (float*)  alloc((size_t)N_NODES * 256 * 4); // h1, then h3
    __bf16*  g_buf  = (__bf16*) alloc((size_t)N_NODES * 512 * 2); // transformed feats (bf16)
    __bf16*  Wt     = (__bf16*) alloc((size_t)512 * 512 * 2);
    float*   asb    = (float*)  alloc((size_t)N_NODES * 4 * 4);
    float*   adb    = (float*)  alloc((size_t)N_NODES * 4 * 4);
    float*   mbuf   = (float*)  alloc((size_t)N_NODES * 4 * 4);
    float*   denom  = (float*)  alloc((size_t)N_NODES * 4 * 4);
    float*   ebuf   = (float*)  alloc((size_t)ET * 4 * 4);
    float*   psums  = (float*)  alloc((size_t)N_GRAPHS * 64 * 4);
    float*   pcnts  = (float*)  alloc((size_t)N_GRAPHS * 4);

    auto fill = [&](float* p, float v, size_t n) {
        fill_f32_kernel<<<cdiv((int)((n + 255) / 256), 1), 256, 0, stream>>>(p, v, n);
    };

    auto run_layer = [&](const float* A, int K, int Nout, int H, int C,
                         const float* W, const float* asrc, const float* adst,
                         const float* bias, float* dst) {
        // 1. W -> Wt (bf16, K-major)
        convert_w_kernel<<<cdiv(K * Nout, 256), 256, 0, stream>>>(W, Wt, K, Nout);
        // 2. g = A @ W  via WMMA
        dim3 gg(cdiv(N_NODES, 128), Nout / 64);
        gemm_wmma_bf16_kernel<<<gg, 256, 0, stream>>>(A, Wt, g_buf, N_NODES, K, Nout);
        // 3. per-node attention coefficients
        alpha_kernel<<<cdiv(N_NODES * H * 32, 256), 256, 0, stream>>>(
            g_buf, asrc, adst, asb, adb, N_NODES, H, C);
        // 4. softmax over incoming edges
        fill(mbuf, -3.402823466e38f, (size_t)N_NODES * H);
        fill(denom, 0.f, (size_t)N_NODES * H);
        edge_logits_kernel<<<cdiv(ET, 256), 256, 0, stream>>>(asb, adb, ei, ebuf, mbuf, H);
        edge_exp_kernel<<<cdiv(ET, 256), 256, 0, stream>>>(ei, mbuf, ebuf, denom, H);
        // 5. weighted scatter
        fill(dst, 0.f, (size_t)N_NODES * H * C);
        int bt = H * C >= 256 ? 256 : H * C;
        edge_aggr_kernel<<<ET, bt, 0, stream>>>(g_buf, ebuf, denom, ei, dst, H, C);
        // 6. bias + relu
        size_t n = (size_t)N_NODES * H * C;
        bias_relu_kernel<<<cdiv((int)((n + 255) / 256), 1), 256, 0, stream>>>(dst, bias, n, H * C);
    };

    // layer 1: 128 -> (4,64) = 256
    run_layer(x,     128, 256, 4, 64,  W1, as1, ad1, b1, o_buf);
    // layer 2: 256 -> (4,128) = 512
    run_layer(o_buf, 256, 512, 4, 128, W2, as2, ad2, b2, f_buf);
    // layer 3: 512 -> (1,64) = 64
    run_layer(f_buf, 512, 64,  1, 64,  W3, as3, ad3, b3, o_buf);

    // global mean pool + FC
    fill(psums, 0.f, (size_t)N_GRAPHS * 64);
    fill(pcnts, 0.f, (size_t)N_GRAPHS);
    pool_kernel<<<N_NODES, 64, 0, stream>>>(o_buf, batch, psums, pcnts);
    fc_kernel<<<cdiv(N_GRAPHS * 10, 128), 128, 0, stream>>>(psums, pcnts, Wfc, bfc, out);
}
